// TwistorLNN_24927990186622
// MI455X (gfx1250) — compile-verified, matrix operand order checked by reading, not run on hardware
//
#include <hip/hip_runtime.h>
#include <hip/hip_bf16.h>

// ---------------- problem constants ----------------
#define T_STEPS 256
#define BATCH   128
#define IN_DIM  512
#define H_DIM   1024
#define OUT_DIM 256

#define DT_C       0.1f
#define TAU_MIN_C  0.01f
#define TAU_MAX_C  1.0f
#define DZDT_MAX_C 10.0f
#define Z_MAX_C    100.0f

// ---------------- CDNA5 WMMA types ----------------
typedef __bf16 bf16_t;
typedef __attribute__((ext_vector_type(16))) __bf16 v16bf;
typedef __attribute__((ext_vector_type(8)))  __bf16 v8bf;
typedef __attribute__((ext_vector_type(8)))  float  v8f;

// float -> bf16 (round to nearest even), no reliance on native __bf16 convert
__device__ __forceinline__ bf16_t f2bf(float f) {
    union { float f; unsigned u; } v; v.f = f;
    unsigned r = v.u + 0x7FFFu + ((v.u >> 16) & 1u);
    unsigned short h = (unsigned short)(r >> 16);
    bf16_t out;
    __builtin_memcpy(&out, &h, 2);
    return out;
}

__device__ __forceinline__ float sigmoidf_(float x) {
    return 1.0f / (1.0f + __expf(-x));
}

// Gather one WMMA operand (16x32 bf16, A-layout per ISA 7.12.2; B uses the
// same gather since B = W^T with W stored row-major [N,K]).
//   lanes 0-15 : row = row0+lane,    K = k0+{0..7}, k0+{16..23}
//   lanes 16-31: row = row0+lane-16, K = k0+{8..15}, k0+{24..31}
__device__ __forceinline__ v16bf load_op(const bf16_t* __restrict__ base,
                                         int ld, int row0, int k0, int lane) {
    int r     = row0 + (lane & 15);
    int kbase = k0 + ((lane & 16) ? 8 : 0);
    const bf16_t* p = base + (size_t)r * ld + kbase;
    v8bf lo = *reinterpret_cast<const v8bf*>(p);        // K = kbase..kbase+7
    v8bf hi = *reinterpret_cast<const v8bf*>(p + 16);   // K = kbase+16..kbase+23
    v16bf out;
#pragma unroll
    for (int i = 0; i < 8; ++i) { out[i] = lo[i]; out[i + 8] = hi[i]; }
    return out;
}

#define WMMA_BF16(a, b, c) \
    __builtin_amdgcn_wmma_f32_16x16x32_bf16(false, (a), false, (b), (short)0, (c), false, false)

// Store one 16x16 f32 tile from the 8-VGPR C layout (ISA 7.12.2)
__device__ __forceinline__ void store_tile(float* __restrict__ C, int ldc,
                                           const float* __restrict__ bias,
                                           int m0, int n0, int lane, v8f acc) {
    int col   = n0 + (lane & 15);
    int rbase = m0 + ((lane & 16) ? 8 : 0);
    float bv = bias ? bias[col] : 0.0f;
#pragma unroll
    for (int v = 0; v < 8; ++v)
        C[(size_t)(rbase + v) * ldc + col] = acc[v] + bv;
}

// One wave computes a 32x32 macro-tile of C[M,N] = A[M,K] * W[N,K]^T (+bias):
// 2x2 register blocking -> 4 operand loads per K-step feed 4 WMMAs.
__device__ __forceinline__ void wmma_tile_2x2(const bf16_t* __restrict__ A, int lda,
                                              const bf16_t* __restrict__ W, int ldw,
                                              float* __restrict__ C, int ldc,
                                              const float* __restrict__ bias,
                                              int m0, int n0, int K, int lane) {
    v8f acc00 = {}, acc01 = {}, acc10 = {}, acc11 = {};
    // per-lane row bases for prefetching the predictable weight stream
    const bf16_t* wpf0 = W + (size_t)(n0 + (lane & 15)) * ldw;
    const bf16_t* wpf1 = W + (size_t)(n0 + 16 + (lane & 15)) * ldw;
    for (int k0 = 0; k0 < K; k0 += 32) {
        if (k0 + 64 < K) {  // pull next macro-K lines toward the WGP (global_prefetch_b8)
            __builtin_prefetch(wpf0 + k0 + 64, 0, 1);
            __builtin_prefetch(wpf1 + k0 + 64, 0, 1);
        }
        v16bf a0 = load_op(A, lda, m0,      k0, lane);
        v16bf a1 = load_op(A, lda, m0 + 16, k0, lane);
        v16bf b0 = load_op(W, ldw, n0,      k0, lane);
        v16bf b1 = load_op(W, ldw, n0 + 16, k0, lane);
        acc00 = WMMA_BF16(a0, b0, acc00);
        acc01 = WMMA_BF16(a0, b1, acc01);
        acc10 = WMMA_BF16(a1, b0, acc10);
        acc11 = WMMA_BF16(a1, b1, acc11);
    }
    store_tile(C, ldc, bias, m0,      n0,      lane, acc00);
    store_tile(C, ldc, bias, m0,      n0 + 16, lane, acc01);
    store_tile(C, ldc, bias, m0 + 16, n0,      lane, acc10);
    store_tile(C, ldc, bias, m0 + 16, n0 + 16, lane, acc11);
}

// ---------------- one-time weight prep: fp32 (optionally masked) -> bf16 ----
__global__ void prep_weights_kernel(const float* __restrict__ w,
                                    const float* __restrict__ mask,
                                    bf16_t* __restrict__ out, int n) {
    int i = blockIdx.x * blockDim.x + threadIdx.x;
    if (i >= n) return;
    float v = w[i];
    if (mask) v *= sigmoidf_(mask[i]);
    out[i] = f2bf(v);
}

// ---------------- per-step kernels ----------------
// 1) activations + x_t conversion + reset norm accumulator
__global__ void step_prep_kernel(const float* __restrict__ zr,
                                 const float* __restrict__ zi,
                                 const float* __restrict__ x_t,
                                 bf16_t* __restrict__ A_r, bf16_t* __restrict__ A_i,
                                 bf16_t* __restrict__ A_m, bf16_t* __restrict__ x_bf,
                                 float* __restrict__ norm_accum) {
    int i = blockIdx.x * blockDim.x + threadIdx.x;
    if (i < BATCH * H_DIM) {
        float r = zr[i], m = zi[i];
        A_r[i] = f2bf(tanhf(r));
        A_i[i] = f2bf(tanhf(m));
        A_m[i] = f2bf(sqrtf(r * r + m * m));
    }
    if (i < BATCH * IN_DIM) x_bf[i] = f2bf(x_t[i]);
    if (i == 0) *norm_accum = 0.0f;
}

// 2) fused 4-way GEMM: blockIdx.z selects {G_r, G_i, G_tau, Ux}
__global__ void step_gemm_kernel(const bf16_t* __restrict__ A_r,
                                 const bf16_t* __restrict__ A_i,
                                 const bf16_t* __restrict__ A_m,
                                 const bf16_t* __restrict__ x_bf,
                                 const bf16_t* __restrict__ Wr,
                                 const bf16_t* __restrict__ Wi,
                                 const bf16_t* __restrict__ Wt,
                                 const bf16_t* __restrict__ U,
                                 float* __restrict__ G_r, float* __restrict__ G_i,
                                 float* __restrict__ G_t, float* __restrict__ Ux) {
    int lane  = threadIdx.x & 31;
    int wave  = threadIdx.x >> 5;
    int ntile = blockIdx.x * 4 + wave;   // 0..31 over H_DIM/32
    int mtile = blockIdx.y;              // 0..3  over BATCH/32
    const bf16_t* A; const bf16_t* W; float* C; int K;
    switch (blockIdx.z) {
        case 0:  A = A_r;  W = Wr; C = G_r; K = H_DIM;  break;
        case 1:  A = A_i;  W = Wi; C = G_i; K = H_DIM;  break;
        case 2:  A = A_m;  W = Wt; C = G_t; K = H_DIM;  break;
        default: A = x_bf; W = U;  C = Ux;  K = IN_DIM; break;
    }
    wmma_tile_2x2(A, K, W, K, C, H_DIM, nullptr, mtile * 32, ntile * 32, K, lane);
}

// 3) pointwise: tau, dz, clip, store dr/di, block-reduce |d| into norm_accum
__global__ void step_pointwise_kernel(const float* __restrict__ G_r,
                                      const float* __restrict__ G_i,
                                      const float* __restrict__ G_t,
                                      const float* __restrict__ Ux,
                                      const float* __restrict__ zr,
                                      const float* __restrict__ zi,
                                      const float* __restrict__ Wr_b,
                                      const float* __restrict__ Wi_b,
                                      const float* __restrict__ U_b,
                                      const float* __restrict__ Wt_b,
                                      const float* __restrict__ tau_bias,
                                      const float* __restrict__ b_real,
                                      const float* __restrict__ b_imag,
                                      float* __restrict__ dr, float* __restrict__ di,
                                      float* __restrict__ norm_accum) {
    int i = blockIdx.x * blockDim.x + threadIdx.x;   // < BATCH*H_DIM
    int h = i & (H_DIM - 1);
    float ux  = Ux[i] + U_b[h];
    float tau = sigmoidf_(G_t[i] + Wt_b[h]) + tau_bias[h];
    tau = fminf(fmaxf(tau, TAU_MIN_C), TAU_MAX_C) + 1e-6f;
    float dzr = -zr[i] + G_r[i] + Wr_b[h] + ux + b_real[h];
    float dzi = -zi[i] + G_i[i] + Wi_b[h] + ux + b_imag[h];
    float r = fminf(fmaxf(dzr / tau, -DZDT_MAX_C), DZDT_MAX_C);
    float m = fminf(fmaxf(dzi / tau, -DZDT_MAX_C), DZDT_MAX_C);
    dr[i] = r; di[i] = m;
    float n = sqrtf(r * r + m * m);

    __shared__ float sdata[256];
    sdata[threadIdx.x] = n;
    __syncthreads();
    for (int s = 128; s > 0; s >>= 1) {
        if (threadIdx.x < s) sdata[threadIdx.x] += sdata[threadIdx.x + s];
        __syncthreads();
    }
    if (threadIdx.x == 0) atomicAdd(norm_accum, sdata[0]);
}

// 4) global rescale + Euler update + clamp + bf16 of new zr for readout
__global__ void step_update_kernel(float* __restrict__ zr, float* __restrict__ zi,
                                   const float* __restrict__ dr,
                                   const float* __restrict__ di,
                                   const float* __restrict__ norm_accum,
                                   bf16_t* __restrict__ A_y) {
    int i = blockIdx.x * blockDim.x + threadIdx.x;   // < BATCH*H_DIM
    float mean  = *norm_accum * (1.0f / (float)(BATCH * H_DIM));
    float half  = DZDT_MAX_C * 0.5f;
    float scale = (mean > half) ? half / (mean + 1e-6f) : 1.0f;
    float r = fminf(fmaxf(zr[i] + DT_C * scale * dr[i], -Z_MAX_C), Z_MAX_C);
    float m = fminf(fmaxf(zi[i] + DT_C * scale * di[i], -Z_MAX_C), Z_MAX_C);
    zr[i] = r; zi[i] = m;
    A_y[i] = f2bf(r);
}

// 5) readout: y_t[B,OUT] = zr_new @ out_w^T + out_b
__global__ void readout_gemm_kernel(const bf16_t* __restrict__ A_y,
                                    const bf16_t* __restrict__ out_w_bf,
                                    const float* __restrict__ out_b,
                                    float* __restrict__ y_t) {
    int lane  = threadIdx.x & 31;
    int wave  = threadIdx.x >> 5;
    int ntile = blockIdx.x * 4 + wave;   // 0..7 over OUT_DIM/32
    int mtile = blockIdx.y;              // 0..3 over BATCH/32
    wmma_tile_2x2(A_y, H_DIM, out_w_bf, H_DIM, y_t, OUT_DIM, out_b,
                  mtile * 32, ntile * 32, H_DIM, lane);
}

// ---------------- host launch ----------------
static inline size_t align256(size_t v) { return (v + 255) & ~(size_t)255; }

extern "C" void kernel_launch(void* const* d_in, const int* in_sizes, int n_in,
                              void* d_out, int out_size, void* d_ws, size_t ws_size,
                              hipStream_t stream) {
    (void)in_sizes; (void)n_in; (void)out_size; (void)ws_size;
    // input order per setup_inputs():
    const float* x        = (const float*)d_in[0];
    const float* Wr_w     = (const float*)d_in[1];
    const float* Wr_b     = (const float*)d_in[2];
    const float* Wi_w     = (const float*)d_in[3];
    const float* Wi_b     = (const float*)d_in[4];
    const float* U_w      = (const float*)d_in[5];
    const float* U_b      = (const float*)d_in[6];
    const float* Wt_w     = (const float*)d_in[7];
    const float* Wt_b     = (const float*)d_in[8];
    const float* mask_r   = (const float*)d_in[9];
    const float* mask_i   = (const float*)d_in[10];
    const float* tau_bias = (const float*)d_in[11];
    const float* b_real   = (const float*)d_in[12];
    const float* b_imag   = (const float*)d_in[13];
    const float* out_w    = (const float*)d_in[14];
    const float* out_b    = (const float*)d_in[15];
    float* y = (float*)d_out;

    // workspace carving (everything stays L2-resident: ~13 MB total)
    char* p = (char*)d_ws;
    auto carve = [&](size_t bytes) { char* r = p; p += align256(bytes); return r; };
    bf16_t* Wr_bf  = (bf16_t*)carve((size_t)H_DIM * H_DIM * 2);
    bf16_t* Wi_bf  = (bf16_t*)carve((size_t)H_DIM * H_DIM * 2);
    bf16_t* Wt_bf  = (bf16_t*)carve((size_t)H_DIM * H_DIM * 2);
    bf16_t* U_bf   = (bf16_t*)carve((size_t)H_DIM * IN_DIM * 2);
    bf16_t* Ow_bf  = (bf16_t*)carve((size_t)OUT_DIM * H_DIM * 2);
    float*  zr     = (float*)carve((size_t)BATCH * H_DIM * 4);
    float*  zi     = (float*)carve((size_t)BATCH * H_DIM * 4);
    bf16_t* A_r    = (bf16_t*)carve((size_t)BATCH * H_DIM * 2);
    bf16_t* A_i    = (bf16_t*)carve((size_t)BATCH * H_DIM * 2);
    bf16_t* A_m    = (bf16_t*)carve((size_t)BATCH * H_DIM * 2);
    bf16_t* A_y    = (bf16_t*)carve((size_t)BATCH * H_DIM * 2);
    bf16_t* x_bf   = (bf16_t*)carve((size_t)BATCH * IN_DIM * 2);
    float*  G_r    = (float*)carve((size_t)BATCH * H_DIM * 4);
    float*  G_i    = (float*)carve((size_t)BATCH * H_DIM * 4);
    float*  G_t    = (float*)carve((size_t)BATCH * H_DIM * 4);
    float*  Ux     = (float*)carve((size_t)BATCH * H_DIM * 4);
    float*  dr     = (float*)carve((size_t)BATCH * H_DIM * 4);
    float*  di     = (float*)carve((size_t)BATCH * H_DIM * 4);
    float*  accum  = (float*)carve(256);

    // zero initial state (graph-capture legal async memset)
    hipMemsetAsync(zr, 0, (size_t)BATCH * H_DIM * 4, stream);
    hipMemsetAsync(zi, 0, (size_t)BATCH * H_DIM * 4, stream);

    // one-time bf16 weight prep (sparsified recurrent weights: W * sigmoid(mask))
    {
        int nHH = H_DIM * H_DIM;
        prep_weights_kernel<<<(nHH + 255) / 256, 256, 0, stream>>>(Wr_w, mask_r, Wr_bf, nHH);
        prep_weights_kernel<<<(nHH + 255) / 256, 256, 0, stream>>>(Wi_w, mask_i, Wi_bf, nHH);
        prep_weights_kernel<<<(nHH + 255) / 256, 256, 0, stream>>>(Wt_w, nullptr, Wt_bf, nHH);
        int nU = H_DIM * IN_DIM;
        prep_weights_kernel<<<(nU + 255) / 256, 256, 0, stream>>>(U_w, nullptr, U_bf, nU);
        int nO = OUT_DIM * H_DIM;
        prep_weights_kernel<<<(nO + 255) / 256, 256, 0, stream>>>(out_w, nullptr, Ow_bf, nO);
    }

    const int nBH = BATCH * H_DIM;                           // 131072
    const dim3 ewGrid((nBH + 255) / 256), ewBlk(256);
    // 4 waves/block, each wave owns a 32x32 macro-tile
    const dim3 gemmGrid(H_DIM / 32 / 4, BATCH / 32, 4), gemmBlk(128);
    const dim3 roGrid(OUT_DIM / 32 / 4, BATCH / 32), roBlk(128);

    for (int t = 0; t < T_STEPS; ++t) {
        const float* x_t = x + (size_t)t * BATCH * IN_DIM;
        float* y_t = y + (size_t)t * BATCH * OUT_DIM;

        step_prep_kernel<<<ewGrid, ewBlk, 0, stream>>>(zr, zi, x_t,
                                                       A_r, A_i, A_m, x_bf, accum);
        step_gemm_kernel<<<gemmGrid, gemmBlk, 0, stream>>>(A_r, A_i, A_m, x_bf,
                                                           Wr_bf, Wi_bf, Wt_bf, U_bf,
                                                           G_r, G_i, G_t, Ux);
        step_pointwise_kernel<<<ewGrid, ewBlk, 0, stream>>>(G_r, G_i, G_t, Ux, zr, zi,
                                                            Wr_b, Wi_b, U_b, Wt_b,
                                                            tau_bias, b_real, b_imag,
                                                            dr, di, accum);
        step_update_kernel<<<ewGrid, ewBlk, 0, stream>>>(zr, zi, dr, di, accum, A_y);
        readout_gemm_kernel<<<roGrid, roBlk, 0, stream>>>(A_y, Ow_bf, out_b, y_t);
    }
}
